// MultiHeadAttention_86105504350851
// MI455X (gfx1250) — matrix operand order repaired and checked by reading, not compiled, not run on hardware
//
#include <hip/hip_runtime.h>

#define KDIM   1024
#define NHEADS 16
#define HEAD   64
#define TSEQ   2048
#define BATCH  2

typedef __attribute__((ext_vector_type(16))) __bf16        v16bf;
typedef __attribute__((ext_vector_type(8)))  float         v8f;
typedef __attribute__((ext_vector_type(4)))  unsigned int  u32x4;

union Frag {
    v16bf           bf;
    u32x4           q[2];
    unsigned short  u[16];
};

__device__ inline unsigned short f2bf(float f) {
    unsigned int u = __float_as_uint(f);
    u += 0x7fffu + ((u >> 16) & 1u);      // round-to-nearest-even
    return (unsigned short)(u >> 16);
}

__device__ inline u32x4 ldg16(const unsigned short* p) {
    return *reinterpret_cast<const u32x4*>(p);
}

__device__ inline v8f wmma_bf16(v16bf a, v16bf b, v8f c) {
    // D = A(16x32 bf16) * B(32x16 bf16) + C(16x16 f32)
    return __builtin_amdgcn_wmma_f32_16x16x32_bf16(
        /*neg_a=*/false, a, /*neg_b=*/false, b,
        /*c_mod=*/(short)0, c, /*reuse_a=*/false, /*reuse_b=*/false);
}

__device__ inline float rmax16(float v) {
    #pragma unroll
    for (int m = 1; m <= 8; m <<= 1) v = fmaxf(v, __shfl_xor(v, m, 32));
    return v;
}

// Async global->LDS copy (CDNA5, tracked by ASYNCcnt), per-lane 16B.
__device__ inline void async_ld_b128(unsigned lds_off, const unsigned short* g) {
    asm volatile("global_load_async_to_lds_b128 %0, %1, off"
                 :: "v"(lds_off), "v"(g) : "memory");
}
__device__ inline void wait_async0() {
    asm volatile("s_wait_asynccnt 0" ::: "memory");
}
__device__ inline unsigned lds_off_of(const void* p) {
    return (unsigned)(unsigned long long)p;   // low 32 bits of flat LDS addr = LDS byte offset
}

// ---------------------------------------------------------------------------
// fp32 -> bf16 elementwise cast
// ---------------------------------------------------------------------------
__global__ void cast_bf16_kernel(const float* __restrict__ in,
                                 unsigned short* __restrict__ out, int n) {
    int i = blockIdx.x * blockDim.x + threadIdx.x;
    if (i < n) out[i] = f2bf(in[i]);
}

// ---------------------------------------------------------------------------
// C[M=4096, N=1024] = A[4096,1024](bf16) * W[1024,1024]^T (bf16)
// MODE 0: out bf16 row-major        (Q, K projections)
// MODE 1: out f32  row-major        (final O @ Wo^T)
// MODE 2: out bf16 transposed per head: Vt[b][h][d][t]   (V projection)
// Block: 256 thr = 8 waves, each wave one 16x64 tile of the same 64-col strip.
// The W strip (identical for all 8 waves) is staged into LDS with async
// copies, double-buffered to overlap with the WMMAs. grid = (32, 16).
// ---------------------------------------------------------------------------
template<int MODE>
__global__ __launch_bounds__(256)
void gemm_xwt_kernel(const unsigned short* __restrict__ A,
                     const unsigned short* __restrict__ W,
                     void* __restrict__ Cout) {
    __shared__ __align__(16) unsigned short bbuf[2][64][72]; // padded rows: 144B stride

    const int tid  = threadIdx.x;
    const int wave = tid >> 5;
    const int lane = tid & 31;
    const int qq   = lane & 15;
    const int hi   = lane >> 4;
    const int m0   = (blockIdx.x * 8 + wave) * 16;
    const int n0   = blockIdx.y * 64;

    const unsigned short* arow = A + (size_t)(m0 + qq) * KDIM;

    // stage W[n0..n0+63][k0..k0+31] into bbuf[s]: 256 thr x 16B = 4KB
    const int sn = tid >> 2;      // 0..63
    const int sc = tid & 3;       // 0..3 (16B chunks of the 64B row)
    const unsigned short* wst = W + (size_t)(n0 + sn) * KDIM + sc * 8;

    v8f acc[4] = {};

    async_ld_b128(lds_off_of(&bbuf[0][sn][sc * 8]), wst);
    wait_async0();
    __syncthreads();

    for (int k0 = 0; k0 < KDIM; k0 += 32) {
        const int cur = (k0 >> 5) & 1;
        if (k0 + 32 < KDIM)
            async_ld_b128(lds_off_of(&bbuf[cur ^ 1][sn][sc * 8]), wst + k0 + 32);

        // A fragment 16x32 from global (per-wave unique rows)
        Frag a;
        a.q[0] = ldg16(arow + k0 + hi * 8);
        a.q[1] = ldg16(arow + k0 + 16 + hi * 8);
        if (k0 + 64 < KDIM) __builtin_prefetch(arow + k0 + 64, 0, 3);

        #pragma unroll
        for (int nt = 0; nt < 4; ++nt) {
            Frag b;
            const unsigned short* bp = &bbuf[cur][nt * 16 + qq][hi * 8];
            b.q[0] = *reinterpret_cast<const u32x4*>(bp);
            b.q[1] = *reinterpret_cast<const u32x4*>(bp + 16);
            acc[nt] = wmma_bf16(a.bf, b.bf, acc[nt]);
        }

        wait_async0();
        __syncthreads();
    }

    // D layout: VGPR j, lanes 0-15 -> (M=j, N=lane); lanes 16-31 -> (M=j+8, N=lane-16)
    if (MODE == 0) {
        unsigned short* C = (unsigned short*)Cout;
        #pragma unroll
        for (int nt = 0; nt < 4; ++nt) {
            int col = n0 + nt * 16 + qq;
            #pragma unroll
            for (int j = 0; j < 8; ++j)
                C[(size_t)(m0 + j + hi * 8) * KDIM + col] = f2bf(acc[nt][j]);
        }
    } else if (MODE == 1) {
        float* C = (float*)Cout;
        #pragma unroll
        for (int nt = 0; nt < 4; ++nt) {
            int col = n0 + nt * 16 + qq;
            #pragma unroll
            for (int j = 0; j < 8; ++j)
                C[(size_t)(m0 + j + hi * 8) * KDIM + col] = acc[nt][j];
        }
    } else {
        // Transposed-per-head store: Vt[((b*16+h)*64+d)*2048 + t]; the 8 rows a
        // lane holds are 8 consecutive tokens -> one packed b128 store per tile.
        unsigned short* Vt = (unsigned short*)Cout;
        const int bb = m0 >> 11;
        const int t0 = (m0 & (TSEQ - 1)) + hi * 8;
        #pragma unroll
        for (int nt = 0; nt < 4; ++nt) {
            int c  = n0 + nt * 16 + qq;
            int hh = c >> 6, dd = c & 63;
            unsigned short tmp[8] __attribute__((aligned(16)));
            #pragma unroll
            for (int j = 0; j < 8; ++j) tmp[j] = f2bf(acc[nt][j]);
            *reinterpret_cast<u32x4*>(
                Vt + ((size_t)((bb * NHEADS + hh) * HEAD + dd)) * TSEQ + t0) =
                *reinterpret_cast<const u32x4*>(tmp);
        }
    }
}

// ---------------------------------------------------------------------------
// Flash-style attention. One wave = one 16-query tile; all 8 waves of a block
// share one (b,h) and stream the same K/V tiles, which are staged into LDS
// with double-buffered async copies. Row-sum of P is computed as one WMMA
// against an all-ones B matrix (only row-max still uses lane shuffles).
// Q,K: bf16 [4096,1024]. Vt: bf16 [b][h][64][2048]. O: bf16 [4096,1024].
// Block: 256 thr = 8 waves. grid = (2048/(8*16)=16, NHEADS, BATCH).
// ---------------------------------------------------------------------------
__global__ __launch_bounds__(256)
void attn_kernel(const unsigned short* __restrict__ Q,
                 const unsigned short* __restrict__ Kt,
                 const unsigned short* __restrict__ Vt,
                 unsigned short* __restrict__ O,
                 float scale) {
    __shared__ __align__(16) unsigned short kbuf[2][32][72]; // 32 keys x 64 dims, padded
    __shared__ __align__(16) unsigned short vbuf[2][64][72]; // 64 dims x 32 keys, padded
    __shared__ __align__(16) unsigned short pbuf[8][16][40]; // per-wave P transpose

    const int tid  = threadIdx.x;
    const int wave = tid >> 5;
    const int lane = tid & 31;
    const int qq   = lane & 15;
    const int hi   = lane >> 4;
    const int b    = blockIdx.z;
    const int h    = blockIdx.y;
    const int q0   = (blockIdx.x * 8 + wave) * 16;
    const int rq   = b * TSEQ + q0;

    const unsigned short* kbase = Kt + (size_t)(b * TSEQ) * KDIM + h * HEAD;
    const unsigned short* vbase = Vt + (size_t)((b * NHEADS + h) * HEAD) * TSEQ;

    // stage mappings (256 threads, 16B each, 4KB per tile)
    const int kn = tid >> 3, kc = tid & 7;            // K: 32 rows x 128B
    const int vn = tid >> 2, vc = tid & 3;            // V: 64 rows x 64B
    const unsigned short* kst = kbase + (size_t)kn * KDIM + kc * 8;
    const unsigned short* vst = vbase + (size_t)vn * TSEQ + vc * 8;

    // Q A-fragments (persist across the whole key loop)
    const unsigned short* qrow = Q + (size_t)(rq + qq) * KDIM + h * HEAD;
    Frag aQ0, aQ1;
    aQ0.q[0] = ldg16(qrow + hi * 8);
    aQ0.q[1] = ldg16(qrow + 16 + hi * 8);
    aQ1.q[0] = ldg16(qrow + 32 + hi * 8);
    aQ1.q[1] = ldg16(qrow + 48 + hi * 8);

    Frag ones;
    #pragma unroll
    for (int i = 0; i < 16; ++i) ones.u[i] = 0x3F80;  // bf16 1.0

    v8f acc[4] = {};
    float mrow[8], lrow[8];
    #pragma unroll
    for (int j = 0; j < 8; ++j) { mrow[j] = -1e30f; lrow[j] = 0.f; }

    // prologue: stage chunk 0 into buffer 0
    async_ld_b128(lds_off_of(&kbuf[0][kn][kc * 8]), kst);
    async_ld_b128(lds_off_of(&vbuf[0][vn][vc * 8]), vst);
    wait_async0();
    __syncthreads();

    for (int kb0 = 0; kb0 < TSEQ; kb0 += 32) {
        const int cur = (kb0 >> 5) & 1;
        if (kb0 + 32 < TSEQ) {
            async_ld_b128(lds_off_of(&kbuf[cur ^ 1][kn][kc * 8]), kst + (size_t)32 * KDIM + kb0 * KDIM);
            async_ld_b128(lds_off_of(&vbuf[cur ^ 1][vn][vc * 8]), vst + kb0 + 32);
        }

        // K B-fragments from LDS: 2 key tiles x 2 k-dim chunks
        Frag bk00, bk01, bk10, bk11;
        {
            const unsigned short* p0 = &kbuf[cur][qq][hi * 8];
            const unsigned short* p1 = &kbuf[cur][16 + qq][hi * 8];
            bk00.q[0] = *reinterpret_cast<const u32x4*>(p0);
            bk00.q[1] = *reinterpret_cast<const u32x4*>(p0 + 16);
            bk01.q[0] = *reinterpret_cast<const u32x4*>(p0 + 32);
            bk01.q[1] = *reinterpret_cast<const u32x4*>(p0 + 48);
            bk10.q[0] = *reinterpret_cast<const u32x4*>(p1);
            bk10.q[1] = *reinterpret_cast<const u32x4*>(p1 + 16);
            bk11.q[0] = *reinterpret_cast<const u32x4*>(p1 + 32);
            bk11.q[1] = *reinterpret_cast<const u32x4*>(p1 + 48);
        }

        v8f z = {};
        v8f S0 = wmma_bf16(aQ0.bf, bk00.bf, z);
        S0     = wmma_bf16(aQ1.bf, bk01.bf, S0);
        v8f S1 = wmma_bf16(aQ0.bf, bk10.bf, z);
        S1     = wmma_bf16(aQ1.bf, bk11.bf, S1);

        // Online softmax over this 16x32 score chunk (max via lane shuffles)
        float corrv[8];
        #pragma unroll
        for (int j = 0; j < 8; ++j) {
            float s0 = S0[j] * scale;
            float s1 = S1[j] * scale;
            float mx   = rmax16(fmaxf(s0, s1));
            float mnew = fmaxf(mrow[j], mx);
            float p0   = __expf(s0 - mnew);
            float p1   = __expf(s1 - mnew);
            corrv[j]   = __expf(mrow[j] - mnew);
            mrow[j]    = mnew;
            acc[0][j] *= corrv[j]; acc[1][j] *= corrv[j];
            acc[2][j] *= corrv[j]; acc[3][j] *= corrv[j];
            // D-layout -> (row, key) layout in LDS
            int row = j + hi * 8;
            pbuf[wave][row][qq]      = f2bf(p0);
            pbuf[wave][row][qq + 16] = f2bf(p1);
        }
        asm volatile("s_wait_dscnt 0" ::: "memory");

        // Re-read P as an A-fragment (16x32, K = 32 keys)
        Frag aP;
        const unsigned short* pr = &pbuf[wave][qq][hi * 8];
        aP.q[0] = *reinterpret_cast<const u32x4*>(pr);
        aP.q[1] = *reinterpret_cast<const u32x4*>(pr + 16);

        // Row-sum of P as one WMMA: D[j] = sum_k P[row(j)][k] for every lane
        v8f sums = wmma_bf16(aP.bf, ones.bf, z);
        #pragma unroll
        for (int j = 0; j < 8; ++j) lrow[j] = lrow[j] * corrv[j] + sums[j];

        // O += P * V using staged transposed V
        #pragma unroll
        for (int nt = 0; nt < 4; ++nt) {
            Frag bv;
            const unsigned short* vr = &vbuf[cur][nt * 16 + qq][hi * 8];
            bv.q[0] = *reinterpret_cast<const u32x4*>(vr);
            bv.q[1] = *reinterpret_cast<const u32x4*>(vr + 16);
            acc[nt] = wmma_bf16(aP.bf, bv.bf, acc[nt]);
        }

        wait_async0();
        __syncthreads();
    }

    // Normalize and store bf16 output tile
    #pragma unroll
    for (int j = 0; j < 8; ++j) {
        float inv = 1.0f / lrow[j];
        int grow = rq + j + hi * 8;
        #pragma unroll
        for (int nt = 0; nt < 4; ++nt)
            O[(size_t)grow * KDIM + h * HEAD + nt * 16 + qq] = f2bf(acc[nt][j] * inv);
    }
}

// ---------------------------------------------------------------------------
extern "C" void kernel_launch(void* const* d_in, const int* in_sizes, int n_in,
                              void* d_out, int out_size, void* d_ws, size_t ws_size,
                              hipStream_t stream) {
    (void)in_sizes; (void)n_in; (void)out_size; (void)ws_size;
    const float* x  = (const float*)d_in[0];
    const float* Wq = (const float*)d_in[1];
    const float* Wk = (const float*)d_in[2];
    const float* Wv = (const float*)d_in[3];
    const float* Wo = (const float*)d_in[4];
    float* out = (float*)d_out;

    const size_t NX = (size_t)BATCH * TSEQ * KDIM;  // 4,194,304
    const size_t NW = (size_t)KDIM * KDIM;          // 1,048,576

    unsigned short* ws  = (unsigned short*)d_ws;
    unsigned short* xb  = ws;           // bf16 x
    unsigned short* wqb = xb  + NX;
    unsigned short* wkb = wqb + NW;
    unsigned short* wvb = wkb + NW;
    unsigned short* wob = wvb + NW;
    unsigned short* Qb  = wob + NW;     // bf16 Q [4096,1024]
    unsigned short* Kb  = Qb  + NX;     // bf16 K [4096,1024]
    unsigned short* Vtb = Kb  + NX;     // bf16 V^T [b][h][64][2048]
    unsigned short* Ob  = Vtb + NX;     // bf16 attention out [4096,1024]
    // total: 24M bf16 elements = 48 MB of workspace

    cast_bf16_kernel<<<(int)((NX + 255) / 256), 256, 0, stream>>>(x,  xb,  (int)NX);
    cast_bf16_kernel<<<(int)((NW + 255) / 256), 256, 0, stream>>>(Wq, wqb, (int)NW);
    cast_bf16_kernel<<<(int)((NW + 255) / 256), 256, 0, stream>>>(Wk, wkb, (int)NW);
    cast_bf16_kernel<<<(int)((NW + 255) / 256), 256, 0, stream>>>(Wv, wvb, (int)NW);
    cast_bf16_kernel<<<(int)((NW + 255) / 256), 256, 0, stream>>>(Wo, wob, (int)NW);

    dim3 gg(32, 16);   // 4096 rows / (8 waves * 16), 1024 cols / 64
    gemm_xwt_kernel<0><<<gg, 256, 0, stream>>>(xb, wqb, Qb);
    gemm_xwt_kernel<0><<<gg, 256, 0, stream>>>(xb, wkb, Kb);
    gemm_xwt_kernel<2><<<gg, 256, 0, stream>>>(xb, wvb, Vtb);

    attn_kernel<<<dim3(16, NHEADS, BATCH), 256, 0, stream>>>(
        Qb, Kb, Vtb, Ob, 0.03125f /* 1/sqrt(1024) */);

    gemm_xwt_kernel<1><<<gg, 256, 0, stream>>>(Ob, wob, out);
}
